// Attention3D_6468220748500
// MI455X (gfx1250) — compile-verified, matrix operand order checked by reading, not run on hardware
//
#include <hip/hip_runtime.h>
#include <hip/hip_bf16.h>

// B=4, C=256, D=H=W=16 -> N=4096. Compute-bound attention: all heavy GEMMs in
// bf16 WMMA (v_wmma_f32_16x16x32_bf16), flash-style attention so the 256MB
// score tensor never touches HBM (K/V tiles are L2-resident: 2MB/batch).

constexpr int CDIM  = 256;
constexpr int NDIM  = 4096;
constexpr int BATCH = 4;

typedef float  v8f   __attribute__((ext_vector_type(8)));
typedef __bf16 bf8v  __attribute__((ext_vector_type(8)));
typedef __bf16 bf16v __attribute__((ext_vector_type(16)));

__device__ __forceinline__ v8f wmma_bf16(bf16v a, bf16v b, v8f c) {
    return __builtin_amdgcn_wmma_f32_16x16x32_bf16(
        /*neg_a=*/false, a, /*neg_b=*/false, b,
        /*c_mod=*/(short)0, c, /*reuse_a=*/false, /*reuse_b=*/false);
}

// Load a 16x32 (MxK) bf16 fragment from a row-major matrix.
// Works for A fragments, and for B fragments when given B^T row-major.
// Lane L: row = L%16, K-halves at 8*(L/16) and 16+8*(L/16)  (ISA 7.12.2).
__device__ __forceinline__ bf16v load_frag(const __bf16* tile, int ld) {
    const int lane = threadIdx.x & 31;
    const __bf16* p = tile + (size_t)(lane & 15) * ld + ((lane >> 4) << 3);
    bf8v a = *reinterpret_cast<const bf8v*>(p);
    bf8v b = *reinterpret_cast<const bf8v*>(p + 16);
    return __builtin_shufflevector(a, b, 0,1,2,3,4,5,6,7,8,9,10,11,12,13,14,15);
}

__device__ __forceinline__ v8f zero8() {
    v8f z = {0.f,0.f,0.f,0.f,0.f,0.f,0.f,0.f};
    return z;
}

// ---------------- weight fp32 -> bf16 ----------------
__global__ void k_cvt_w(const float* __restrict__ wq, const float* __restrict__ wk,
                        const float* __restrict__ wv, const float* __restrict__ wp,
                        __bf16* __restrict__ o) {
    int i = blockIdx.x * 256 + threadIdx.x;          // 65536 threads
    o[i]          = (__bf16)wq[i];
    o[i +  65536] = (__bf16)wk[i];
    o[i + 131072] = (__bf16)wv[i];
    o[i + 196608] = (__bf16)wp[i];
}

// ---------------- x (B,C,N) f32 -> xT (B,N,C) bf16, LDS-tiled ----------------
__global__ void k_transpose(const float* __restrict__ x, __bf16* __restrict__ xT) {
    __shared__ float tile[32][33];
    int b  = blockIdx.z;
    int n0 = blockIdx.x * 32;
    int c0 = blockIdx.y * 32;
    int tx = threadIdx.x & 31;
    int ty = threadIdx.x >> 5;                       // 0..7
    const float* xp = x + ((size_t)b * CDIM + c0) * NDIM + n0;
#pragma unroll
    for (int k = 0; k < 4; k++)
        tile[ty + 8*k][tx] = xp[(size_t)(ty + 8*k) * NDIM + tx];   // tile[c][n]
    __syncthreads();
    __bf16* op = xT + ((size_t)b * NDIM + n0) * CDIM + c0;
#pragma unroll
    for (int k = 0; k < 4; k++)
        op[(size_t)(ty + 8*k) * CDIM + tx] = (__bf16)tile[tx][ty + 8*k];
}

// ---------------- generic WMMA GEMM: Out(MxNout) = A(MxK) * Bm(NoutxK)^T + bias
// 8 waves/block, each wave a 32x32 tile; block tile 128x64.
__global__ void k_gemm_bias(const __bf16* __restrict__ A, const __bf16* __restrict__ Bm,
                            const float* __restrict__ bias, __bf16* __restrict__ Out,
                            int K, int Nout,
                            long strideA, long strideB, long strideO, int bias_row) {
    int bz   = blockIdx.z;
    const __bf16* Abase = A + (size_t)bz * strideA;
    const __bf16* Bbase = Bm + (size_t)bz * strideB;
    __bf16*       Obase = Out + (size_t)bz * strideO;
    int wave = threadIdx.x >> 5;
    int lane = threadIdx.x & 31;
    int m0 = blockIdx.x * 128 + (wave >> 1) * 32;
    int n0 = blockIdx.y *  64 + (wave &  1) * 32;
    v8f acc00 = zero8(), acc01 = zero8(), acc10 = zero8(), acc11 = zero8();
    for (int k0 = 0; k0 < K; k0 += 32) {
        bf16v a0 = load_frag(Abase + (size_t)m0        * K + k0, K);
        bf16v a1 = load_frag(Abase + (size_t)(m0 + 16) * K + k0, K);
        bf16v b0 = load_frag(Bbase + (size_t)n0        * K + k0, K);
        bf16v b1 = load_frag(Bbase + (size_t)(n0 + 16) * K + k0, K);
        acc00 = wmma_bf16(a0, b0, acc00);
        acc01 = wmma_bf16(a0, b1, acc01);
        acc10 = wmma_bf16(a1, b0, acc10);
        acc11 = wmma_bf16(a1, b1, acc11);
    }
    int col = lane & 15, h = lane >> 4;
#pragma unroll
    for (int fm = 0; fm < 2; fm++) {
#pragma unroll
        for (int fn = 0; fn < 2; fn++) {
            v8f acc = fm ? (fn ? acc11 : acc10) : (fn ? acc01 : acc00);
#pragma unroll
            for (int v = 0; v < 8; v++) {
                int m = m0 + fm * 16 + v + 8 * h;
                int n = n0 + fn * 16 + col;
                float val = acc[v] + (bias_row ? bias[m] : bias[n]);
                Obase[(size_t)m * Nout + n] = (__bf16)val;
            }
        }
    }
}

// ---------------- flash attention: per wave = 16 query rows ----------------
// Qt,Kt: (B,N,C) bf16; Vm: (B,C,N) bf16; attnT out: (B,N,C) bf16
__global__ void k_attn(const __bf16* __restrict__ Qt, const __bf16* __restrict__ Kt,
                       const __bf16* __restrict__ Vm, __bf16* __restrict__ attnT) {
    __shared__ __align__(16) __bf16 plds[8][16 * 32];   // P staging, per wave
    int b    = blockIdx.y;
    int wave = threadIdx.x >> 5;
    int lane = threadIdx.x & 31;
    int col  = lane & 15, h = lane >> 4;
    int i0   = blockIdx.x * 128 + wave * 16;
    const __bf16* Q  = Qt + (size_t)b * NDIM * CDIM;
    const __bf16* Kp = Kt + (size_t)b * NDIM * CDIM;
    const __bf16* Vp = Vm + (size_t)b * CDIM * NDIM;
    __bf16* pw = &plds[wave][0];

    bf16v qf[8];
#pragma unroll
    for (int kc = 0; kc < 8; kc++)
        qf[kc] = load_frag(Q + (size_t)i0 * CDIM + kc * 32, CDIM);

    v8f o[16];
#pragma unroll
    for (int cf = 0; cf < 16; cf++) o[cf] = zero8();
    float mr[8], lr[8];
#pragma unroll
    for (int v = 0; v < 8; v++) { mr[v] = -1.0e30f; lr[v] = 0.0f; }

    const float scale = 0.0625f;   // C^-0.5, C=256
    for (int j0 = 0; j0 < NDIM; j0 += 32) {
        if (j0 + 32 < NDIM)
            __builtin_prefetch(Kp + (size_t)(j0 + 32) * CDIM, 0, 0);
        // S tiles: (16 i) x (32 j), K-loop over channels
        v8f s0 = zero8(), s1 = zero8();
#pragma unroll
        for (int kc = 0; kc < 8; kc++) {
            bf16v k0f = load_frag(Kp + (size_t)j0        * CDIM + kc * 32, CDIM);
            bf16v k1f = load_frag(Kp + (size_t)(j0 + 16) * CDIM + kc * 32, CDIM);
            s0 = wmma_bf16(qf[kc], k0f, s0);
            s1 = wmma_bf16(qf[kc], k1f, s1);
        }
        // online softmax per row (row = v + 8*h in D layout)
#pragma unroll
        for (int v = 0; v < 8; v++) {
            float a  = s0[v] * scale;
            float c2 = s1[v] * scale;
            float mx = fmaxf(a, c2);
#pragma unroll
            for (int off = 1; off < 16; off <<= 1)
                mx = fmaxf(mx, __shfl_xor(mx, off, 32));
            float mnew = fmaxf(mr[v], mx);
            float corr = __expf(mr[v] - mnew);
            mr[v] = mnew;
            float p0 = __expf(a  - mnew);
            float p1 = __expf(c2 - mnew);
            float rs = p0 + p1;
#pragma unroll
            for (int off = 1; off < 16; off <<= 1)
                rs += __shfl_xor(rs, off, 32);
            lr[v] = lr[v] * corr + rs;
#pragma unroll
            for (int cf = 0; cf < 16; cf++) o[cf][v] *= corr;
            int r = v + 8 * h;
            pw[r * 32 + col]      = (__bf16)p0;   // P row-major 16x32 in LDS
            pw[r * 32 + 16 + col] = (__bf16)p1;
        }
        // Reload P as an A fragment (wave-private LDS; in-wave DS ordering)
        bf16v pf = load_frag(pw, 32);
        // O(i,c) += P(i,j) * V(c,j)^T  -> B^T = V row-major (C,N)
#pragma unroll
        for (int cf = 0; cf < 16; cf++) {
            bf16v vf = load_frag(Vp + (size_t)(cf * 16) * NDIM + j0, NDIM);
            o[cf] = wmma_bf16(pf, vf, o[cf]);
        }
    }
    // normalize and store attnT (B,N,C)
    __bf16* Ob = attnT + (size_t)b * NDIM * CDIM;
#pragma unroll
    for (int v = 0; v < 8; v++) {
        float inv = 1.0f / lr[v];
        int i = i0 + v + 8 * h;
#pragma unroll
        for (int cf = 0; cf < 16; cf++)
            Ob[(size_t)i * CDIM + cf * 16 + col] = (__bf16)(o[cf][v] * inv);
    }
}

// ---------------- out = x + wp*attn + bp (fp32), fused residual ----------------
__global__ void k_proj_out(const __bf16* __restrict__ wpb, const __bf16* __restrict__ attnT,
                           const float* __restrict__ bp, const float* __restrict__ x,
                           float* __restrict__ out) {
    int b    = blockIdx.z;
    int wave = threadIdx.x >> 5;
    int lane = threadIdx.x & 31;
    int m0 = blockIdx.x * 128 + (wave >> 1) * 32;    // output channel o
    int n0 = blockIdx.y *  64 + (wave &  1) * 32;    // voxel n
    const __bf16* Bt = attnT + (size_t)b * NDIM * CDIM;  // (N,C) = B^T
    v8f acc00 = zero8(), acc01 = zero8(), acc10 = zero8(), acc11 = zero8();
    for (int k0 = 0; k0 < CDIM; k0 += 32) {
        bf16v a0 = load_frag(wpb + (size_t)m0        * CDIM + k0, CDIM);
        bf16v a1 = load_frag(wpb + (size_t)(m0 + 16) * CDIM + k0, CDIM);
        bf16v b0 = load_frag(Bt  + (size_t)n0        * CDIM + k0, CDIM);
        bf16v b1 = load_frag(Bt  + (size_t)(n0 + 16) * CDIM + k0, CDIM);
        acc00 = wmma_bf16(a0, b0, acc00);
        acc01 = wmma_bf16(a0, b1, acc01);
        acc10 = wmma_bf16(a1, b0, acc10);
        acc11 = wmma_bf16(a1, b1, acc11);
    }
    int col = lane & 15, h = lane >> 4;
#pragma unroll
    for (int fm = 0; fm < 2; fm++) {
#pragma unroll
        for (int fn = 0; fn < 2; fn++) {
            v8f acc = fm ? (fn ? acc11 : acc10) : (fn ? acc01 : acc00);
#pragma unroll
            for (int v = 0; v < 8; v++) {
                int m = m0 + fm * 16 + v + 8 * h;
                int n = n0 + fn * 16 + col;
                size_t idx = ((size_t)b * CDIM + m) * NDIM + n;
                out[idx] = x[idx] + acc[v] + bp[m];
            }
        }
    }
}

extern "C" void kernel_launch(void* const* d_in, const int* in_sizes, int n_in,
                              void* d_out, int out_size, void* d_ws, size_t ws_size,
                              hipStream_t stream) {
    const float* x  = (const float*)d_in[0];
    const float* wq = (const float*)d_in[1];
    const float* bq = (const float*)d_in[2];
    const float* wk = (const float*)d_in[3];
    const float* bk = (const float*)d_in[4];
    const float* wv = (const float*)d_in[5];
    const float* bv = (const float*)d_in[6];
    const float* wp = (const float*)d_in[7];
    const float* bp = (const float*)d_in[8];
    float* out = (float*)d_out;

    const size_t WSZ = (size_t)CDIM * CDIM;      // 65536
    const size_t XSZ = (size_t)BATCH * NDIM * CDIM; // 4*4096*256
    __bf16* wqb   = (__bf16*)d_ws;
    __bf16* wkb   = wqb + WSZ;
    __bf16* wvb   = wqb + 2 * WSZ;
    __bf16* wpb   = wqb + 3 * WSZ;
    __bf16* xT    = wqb + 4 * WSZ;
    __bf16* Qt    = xT   + XSZ;
    __bf16* Ktb   = Qt   + XSZ;
    __bf16* Vm    = Ktb  + XSZ;
    __bf16* attnT = Vm   + XSZ;

    // 1) weights -> bf16
    k_cvt_w<<<256, 256, 0, stream>>>(wq, wk, wv, wp, wqb);
    // 2) x (B,C,N) f32 -> xT (B,N,C) bf16
    k_transpose<<<dim3(NDIM / 32, CDIM / 32, BATCH), 256, 0, stream>>>(x, xT);
    // 3) projections. Qt/Kt (B,N,C): A=xT (batched), Bm=weight; bias per column.
    k_gemm_bias<<<dim3(NDIM / 128, CDIM / 64, BATCH), 256, 0, stream>>>(
        xT, wqb, bq, Qt, CDIM, CDIM, (long)NDIM * CDIM, 0L, (long)NDIM * CDIM, 0);
    k_gemm_bias<<<dim3(NDIM / 128, CDIM / 64, BATCH), 256, 0, stream>>>(
        xT, wkb, bk, Ktb, CDIM, CDIM, (long)NDIM * CDIM, 0L, (long)NDIM * CDIM, 0);
    //    V (B,C,N): A=wv, Bm=xT (batched); bias per row.
    k_gemm_bias<<<dim3(CDIM / 128, NDIM / 64, BATCH), 256, 0, stream>>>(
        wvb, xT, bv, Vm, CDIM, NDIM, 0L, (long)NDIM * CDIM, (long)CDIM * NDIM, 1);
    // 4) flash attention -> attnT (B,N,C)
    k_attn<<<dim3(NDIM / 128, BATCH), 256, 0, stream>>>(Qt, Ktb, Vm, attnT);
    // 5) output projection + bias + residual (fp32)
    k_proj_out<<<dim3(CDIM / 128, NDIM / 64, BATCH), 256, 0, stream>>>(wpb, attnT, bp, x, out);
}
